// MultiHeadAttention_53541062312292
// MI455X (gfx1250) — matrix operand order repaired and checked by reading, not compiled, not run on hardware
//
#include <hip/hip_runtime.h>

#define EMB   2048
#define HEADS 16
#define HD    128
#define BATCH 2
#define QLEN  1024
#define TC    3072
#define KLEN  4096   // TC + QLEN

typedef __attribute__((ext_vector_type(16))) _Float16 v16h;
typedef __attribute__((ext_vector_type(8)))  float    v8f;

// ---------------------------------------------------------------------------
// Kernel 1: copy caches into the t<TC part of hk/hv output regions.
// ---------------------------------------------------------------------------
__global__ void copy_caches_kernel(const float4* __restrict__ ck,
                                   const float4* __restrict__ cv,
                                   float4* __restrict__ hk,
                                   float4* __restrict__ hv) {
  const long ROW4  = EMB / 4;               // 512 float4 per token row (H*D)
  const long PER_B = (long)TC * ROW4;
  const long TOTAL = (long)BATCH * PER_B;
  for (long i = (long)blockIdx.x * blockDim.x + threadIdx.x; i < TOTAL;
       i += (long)gridDim.x * blockDim.x) {
    long b = i / PER_B;
    long r = i - b * PER_B;
    long dst = b * (long)KLEN * ROW4 + r;
    hk[dst] = ck[i];
    hv[dst] = cv[i];
  }
}

// ---------------------------------------------------------------------------
// Kernel 2: projection GEMM  y[m,n] = sum_k x[m,k]*W[n,k] + b[n]
// SCATTER==0: write f16 into dstH (query, workspace)
// SCATTER==1: scatter f32 into dstF at [b, TC+t, :] (key/value into hk/hv)
// ---------------------------------------------------------------------------
#define TM  128
#define TN  128
#define TKK 64
#define LDA (TKK + 8)   // 72 halfs, padded

template <int SCATTER>
__global__ __launch_bounds__(256)
void qkv_gemm_kernel(const float* __restrict__ X,
                     const float* __restrict__ W,
                     const float* __restrict__ bias,
                     float* __restrict__ dstF,
                     _Float16* __restrict__ dstH) {
  __shared__ _Float16 As[TM][LDA];
  __shared__ _Float16 Bs[TN][LDA];

  const int tid  = threadIdx.x;
  const int wave = tid >> 5;
  const int lane = tid & 31;
  const int half = lane >> 4;
  const int l16  = lane & 15;
  const int wm   = wave >> 2;   // 0..1  -> 64 rows each
  const int wn   = wave & 3;    // 0..3  -> 32 cols each
  const int m0   = blockIdx.x * TM;
  const int n0   = blockIdx.y * TN;

  v8f acc[4][2] = {};

  const int srow  = tid >> 1;          // 0..127
  const int skoff = (tid & 1) * 32;    // 0 or 32

  for (int k0 = 0; k0 < EMB; k0 += TKK) {
    const float* pa = X + (size_t)(m0 + srow) * EMB + k0 + skoff;
    const float* pb = W + (size_t)(n0 + srow) * EMB + k0 + skoff;
#pragma unroll
    for (int j = 0; j < 32; j += 4) {
      float4 a = *(const float4*)(pa + j);
      float4 b = *(const float4*)(pb + j);
      As[srow][skoff + j + 0] = (_Float16)a.x;
      As[srow][skoff + j + 1] = (_Float16)a.y;
      As[srow][skoff + j + 2] = (_Float16)a.z;
      As[srow][skoff + j + 3] = (_Float16)a.w;
      Bs[srow][skoff + j + 0] = (_Float16)b.x;
      Bs[srow][skoff + j + 1] = (_Float16)b.y;
      Bs[srow][skoff + j + 2] = (_Float16)b.z;
      Bs[srow][skoff + j + 3] = (_Float16)b.w;
    }
    __syncthreads();

#pragma unroll
    for (int ks = 0; ks < TKK; ks += 32) {
      v16h afrag[4], bfrag[2];
#pragma unroll
      for (int i = 0; i < 4; i++) {
        const _Float16* ap = &As[wm * 64 + i * 16 + l16][ks];
#pragma unroll
        for (int j = 0; j < 8; j++) {
          afrag[i][j]     = ap[half * 8 + j];        // K = 8*half + j
          afrag[i][j + 8] = ap[16 + half * 8 + j];   // K = 16 + 8*half + j
        }
      }
#pragma unroll
      for (int i = 0; i < 2; i++) {
        const _Float16* bp = &Bs[wn * 32 + i * 16 + l16][ks];
#pragma unroll
        for (int j = 0; j < 8; j++) {
          bfrag[i][j]     = bp[half * 8 + j];
          bfrag[i][j + 8] = bp[16 + half * 8 + j];
        }
      }
#pragma unroll
      for (int i = 0; i < 4; i++)
#pragma unroll
        for (int j = 0; j < 2; j++)
          acc[i][j] = __builtin_amdgcn_wmma_f32_16x16x32_f16(
              false, afrag[i], false, bfrag[j], (short)0, acc[i][j], false,
              false);
    }
    __syncthreads();
  }

  // epilogue: C layout -> lane holds (M = r + 8*half, N = l16) per subtile
#pragma unroll
  for (int i = 0; i < 4; i++) {
#pragma unroll
    for (int j = 0; j < 2; j++) {
      const int col = n0 + wn * 32 + j * 16 + l16;
      const float bb = bias[col];
#pragma unroll
      for (int r = 0; r < 8; r++) {
        const int m = m0 + wm * 64 + i * 16 + half * 8 + r;
        const float v = acc[i][j][r] + bb;
        if constexpr (SCATTER == 0) {
          dstH[(size_t)m * EMB + col] = (_Float16)v;
        } else {
          const int b = m >> 10;             // m / QLEN
          const int t = m & (QLEN - 1);
          dstF[((size_t)b * KLEN + TC + t) * EMB + col] = v;
        }
      }
    }
  }
}

// ---------------------------------------------------------------------------
// Kernel 3: flash-style attention (no scale, no mask, matching reference).
// One workgroup = 128 Q rows of one (b,h). 8 waves, 16 rows each.
// K chunk of 64; V staged transposed so B-fragments are contiguous b128 loads.
// Q staging tile shares LDS (union) with the per-chunk buffers.
// ---------------------------------------------------------------------------
#define AQ  128
#define AKC 64
#define LQ  (HD + 8)     // 136 halfs
#define LK  (HD + 8)     // 136 halfs
#define LV  (AKC + 8)    // 72 halfs
#define LP  (AKC + 8)    // 72 halfs

__global__ __launch_bounds__(256)
void attention_kernel(const _Float16* __restrict__ Qm,   // [B*QLEN, EMB] f16
                      const float* __restrict__ hk,      // [B,KLEN,H,HD] f32
                      const float* __restrict__ hv,
                      float* __restrict__ out) {         // [B,QLEN,EMB] f32
  __shared__ __align__(16) union {
    _Float16 q[AQ][LQ];                 // 34816 B (staging only)
    struct {
      _Float16 k[AKC][LK];              // 17408 B : k[t][d]
      _Float16 vT[HD][LV];              // 18432 B : vT[d][t]
      _Float16 p[8][16][LP];            // 18432 B : per-wave P tiles
    } s;
  } sm;                                 // 54272 B total

  const int tid  = threadIdx.x;
  const int wave = tid >> 5;
  const int lane = tid & 31;
  const int half = lane >> 4;
  const int l16  = lane & 15;
  const int h  = blockIdx.y;
  const int b  = blockIdx.z;
  const int q0 = blockIdx.x * AQ;

  // ---- stage Q tile (f16, vectorized 16B copies; padded rows stay aligned)
  {
    const int row = tid >> 1;
    const int c0  = (tid & 1) * 64;
    const uint4* src = (const uint4*)(Qm +
        ((size_t)(b * QLEN + q0 + row)) * EMB + h * HD + c0);
    uint4* dst = (uint4*)&sm.q[row][c0];
#pragma unroll
    for (int j = 0; j < 8; j++) dst[j] = src[j];
  }
  __syncthreads();

  // ---- preload Q A-fragments (rows wave*16 + l16), 4 K-steps over d
  v16h qfrag[4];
#pragma unroll
  for (int d0 = 0; d0 < 4; d0++) {
    const _Float16* qp = &sm.q[wave * 16 + l16][d0 * 32];
#pragma unroll
    for (int j = 0; j < 8; j++) {
      qfrag[d0][j]     = qp[half * 8 + j];
      qfrag[d0][j + 8] = qp[16 + half * 8 + j];
    }
  }
  __syncthreads();   // Q reads done; union memory may now be reused

  float mrow[8], lrow[8];
  v8f oacc[8] = {};
#pragma unroll
  for (int r = 0; r < 8; r++) { mrow[r] = -1e30f; lrow[r] = 0.0f; }

  for (int t0 = 0; t0 < KLEN; t0 += AKC) {
    // ---- stage K (row-major) and V (transposed), f32 -> f16
    {
      const int trow = tid >> 2;          // 0..63
      const int c0   = (tid & 3) * 32;    // 0,32,64,96
      const size_t base =
          (((size_t)b * KLEN + t0 + trow) * HEADS + h) * HD + c0;
      const float* pk = hk + base;
      const float* pv = hv + base;
#pragma unroll
      for (int j = 0; j < 32; j += 4) {
        float4 kk = *(const float4*)(pk + j);
        float4 vv = *(const float4*)(pv + j);
        sm.s.k[trow][c0 + j + 0] = (_Float16)kk.x;
        sm.s.k[trow][c0 + j + 1] = (_Float16)kk.y;
        sm.s.k[trow][c0 + j + 2] = (_Float16)kk.z;
        sm.s.k[trow][c0 + j + 3] = (_Float16)kk.w;
        sm.s.vT[c0 + j + 0][trow] = (_Float16)vv.x;
        sm.s.vT[c0 + j + 1][trow] = (_Float16)vv.y;
        sm.s.vT[c0 + j + 2][trow] = (_Float16)vv.z;
        sm.s.vT[c0 + j + 3][trow] = (_Float16)vv.w;
      }
      if (t0 + AKC < KLEN) {
        const size_t nbase =
            (((size_t)b * KLEN + t0 + AKC + trow) * HEADS + h) * HD + c0;
        __builtin_prefetch(hk + nbase, 0, 0);   // global_prefetch_b8
        __builtin_prefetch(hv + nbase, 0, 0);
      }
    }
    __syncthreads();

    // ---- S(16x64) = Q(16x128) @ K^T : 4 col-subtiles, K-dim chained over d
    v8f sacc[4] = {};
#pragma unroll
    for (int sub = 0; sub < 4; sub++) {
#pragma unroll
      for (int d0 = 0; d0 < 4; d0++) {
        v16h kf;
        const _Float16* kp = &sm.s.k[sub * 16 + l16][d0 * 32];
#pragma unroll
        for (int j = 0; j < 8; j++) {
          kf[j]     = kp[half * 8 + j];
          kf[j + 8] = kp[16 + half * 8 + j];
        }
        sacc[sub] = __builtin_amdgcn_wmma_f32_16x16x32_f16(
            false, qfrag[d0], false, kf, (short)0, sacc[sub], false, false);
      }
    }

    // ---- online softmax across the 64 new columns
#pragma unroll
    for (int r = 0; r < 8; r++) {
      float v0 = sacc[0][r], v1 = sacc[1][r];
      float v2 = sacc[2][r], v3 = sacc[3][r];
      float cmax = fmaxf(fmaxf(v0, v1), fmaxf(v2, v3));
#pragma unroll
      for (int off = 1; off < 16; off <<= 1)
        cmax = fmaxf(cmax, __shfl_xor(cmax, off, 16));
      const float mnew  = fmaxf(mrow[r], cmax);
      const float scale = __expf(mrow[r] - mnew);
      const float p0 = __expf(v0 - mnew);
      const float p1 = __expf(v1 - mnew);
      const float p2 = __expf(v2 - mnew);
      const float p3 = __expf(v3 - mnew);
      float psum = (p0 + p1) + (p2 + p3);
#pragma unroll
      for (int off = 1; off < 16; off <<= 1)
        psum += __shfl_xor(psum, off, 16);
      lrow[r] = lrow[r] * scale + psum;
      mrow[r] = mnew;
#pragma unroll
      for (int d = 0; d < 8; d++) oacc[d][r] = oacc[d][r] * scale;
      // C-layout -> per-wave LDS tile (A-layout pickup below)
      _Float16* pr = &sm.s.p[wave][half * 8 + r][0];
      pr[l16]      = (_Float16)p0;
      pr[16 + l16] = (_Float16)p1;
      pr[32 + l16] = (_Float16)p2;
      pr[48 + l16] = (_Float16)p3;
    }
    __syncthreads();

    // ---- O(16x128) += P(16x64) @ V(64x128) : 2 K-blocks x 8 d-subtiles
#pragma unroll
    for (int kb = 0; kb < 2; kb++) {
      v16h pfrag;
      const _Float16* pp = &sm.s.p[wave][l16][kb * 32];
#pragma unroll
      for (int j = 0; j < 8; j++) {
        pfrag[j]     = pp[half * 8 + j];
        pfrag[j + 8] = pp[16 + half * 8 + j];
      }
#pragma unroll
      for (int d = 0; d < 8; d++) {
        v16h vf;
        const _Float16* vp = &sm.s.vT[d * 16 + l16][kb * 32];
#pragma unroll
        for (int j = 0; j < 8; j++) {
          vf[j]     = vp[half * 8 + j];
          vf[j + 8] = vp[16 + half * 8 + j];
        }
        oacc[d] = __builtin_amdgcn_wmma_f32_16x16x32_f16(
            false, pfrag, false, vf, (short)0, oacc[d], false, false);
      }
    }
    __syncthreads();
  }

  // ---- epilogue: normalize and store
#pragma unroll
  for (int d = 0; d < 8; d++) {
#pragma unroll
    for (int r = 0; r < 8; r++) {
      const int qrow = q0 + wave * 16 + half * 8 + r;
      const int col  = h * HD + d * 16 + l16;
      out[((size_t)b * QLEN + qrow) * EMB + col] = oacc[d][r] / lrow[r];
    }
  }
}

// ---------------------------------------------------------------------------
extern "C" void kernel_launch(void* const* d_in, const int* in_sizes, int n_in,
                              void* d_out, int out_size, void* d_ws,
                              size_t ws_size, hipStream_t stream) {
  (void)in_sizes; (void)n_in; (void)out_size; (void)ws_size;
  const float* q  = (const float*)d_in[0];
  const float* ck = (const float*)d_in[1];
  const float* cv = (const float*)d_in[2];
  const float* Wq = (const float*)d_in[3];
  const float* bq = (const float*)d_in[4];
  const float* Wk = (const float*)d_in[5];
  const float* bk = (const float*)d_in[6];
  const float* Wv = (const float*)d_in[7];
  const float* bv = (const float*)d_in[8];

  float* out = (float*)d_out;
  float* hk  = out + (size_t)BATCH * QLEN * EMB;   // [B,KLEN,H,HD]
  float* hv  = hk + (size_t)BATCH * KLEN * EMB;
  _Float16* qproj = (_Float16*)d_ws;               // [B*QLEN, EMB] f16

  copy_caches_kernel<<<2048, 256, 0, stream>>>(
      (const float4*)ck, (const float4*)cv, (float4*)hk, (float4*)hv);

  dim3 gg((BATCH * QLEN) / TM, EMB / TN);          // (16,16)
  qkv_gemm_kernel<0><<<gg, 256, 0, stream>>>(q, Wq, bq, nullptr, qproj);
  qkv_gemm_kernel<1><<<gg, 256, 0, stream>>>(q, Wk, bk, hk, nullptr);
  qkv_gemm_kernel<1><<<gg, 256, 0, stream>>>(q, Wv, bv, hv, nullptr);

  dim3 ga(QLEN / AQ, HEADS, BATCH);                // (8,16,2)
  attention_kernel<<<ga, 256, 0, stream>>>(qproj, hk, hv, out);
}
